// SoftDecisionTree_90039694393610
// MI455X (gfx1250) — matrix operand order repaired and checked by reading, not compiled, not run on hardware
//
#include <hip/hip_runtime.h>

typedef _Float16 h8   __attribute__((ext_vector_type(8)));
typedef _Float16 v16h __attribute__((ext_vector_type(16)));
typedef float    v8f  __attribute__((ext_vector_type(8)));
typedef float    f4   __attribute__((ext_vector_type(4)));

union Frag { v16h v; h8 h[2]; };

#define BATCH_N 8192
#define IN_F    3072
#define N_NODE  1023
#define N_LEAF  1024
#define N_CLS   1000
#define N_PAD   1024

#define BM  128
#define BN  128
#define BK  32
#define LDT 40   // padded LDS row stride in f16 elements (80B -> staggered banks)

// ---------------------------------------------------------------------------
// Kernel 1: ps = sigmoid(xs @ W^T + b)   M=8192 N=1023(pad 1024) K=3072
// f32 inputs converted to f16 in registers, WMMA f32 accumulate.
// ---------------------------------------------------------------------------
__global__ __launch_bounds__(256)
void sdt_gemm1_sigmoid(const float* __restrict__ xs, const float* __restrict__ W,
                       const float* __restrict__ bias, float* __restrict__ ps)
{
    __shared__ _Float16 At[BM * LDT];
    __shared__ _Float16 Bt[BN * LDT];

    const int tid  = threadIdx.x;
    const int lane = tid & 31;
    const int wid  = tid >> 5;
    const int bm   = blockIdx.x * BM;
    const int bn   = blockIdx.y * BN;
    const int wm   = (wid & 3) * 32;   // wave M offset inside tile
    const int wn   = (wid >> 2) * 64;  // wave N offset inside tile

    const int lrow  = tid >> 1;        // 0..127 : tile row this thread stages
    const int lhalf = (tid & 1) * 16;  // 0 / 16 : K half it stages

    v8f acc[2][4] = {};

    const float* aSrc = xs + (size_t)(bm + lrow) * IN_F + lhalf;
    const int    wrow = bn + lrow;
    const float* bSrc = W + (size_t)wrow * IN_F + lhalf;

    for (int k0 = 0; k0 < IN_F; k0 += BK) {
        // ---- stage: global f32 loads (issued before barrier to overlap) ----
        f4 a0 = *(const f4*)(aSrc + k0);
        f4 a1 = *(const f4*)(aSrc + k0 + 4);
        f4 a2 = *(const f4*)(aSrc + k0 + 8);
        f4 a3 = *(const f4*)(aSrc + k0 + 12);
        f4 b0 = {0.f, 0.f, 0.f, 0.f}, b1 = b0, b2 = b0, b3 = b0;
        if (wrow < N_NODE) {
            b0 = *(const f4*)(bSrc + k0);
            b1 = *(const f4*)(bSrc + k0 + 4);
            b2 = *(const f4*)(bSrc + k0 + 8);
            b3 = *(const f4*)(bSrc + k0 + 12);
        }
        if (k0 + BK < IN_F) {
            __builtin_prefetch(aSrc + k0 + BK, 0, 1);   // global_prefetch_b8
            __builtin_prefetch(bSrc + k0 + BK, 0, 1);
        }

        __syncthreads();   // previous tile fully consumed
        h8 pa0, pa1, pb0, pb1;
        #pragma unroll
        for (int j = 0; j < 4; ++j) {
            pa0[j] = (_Float16)a0[j]; pa0[j + 4] = (_Float16)a1[j];
            pa1[j] = (_Float16)a2[j]; pa1[j + 4] = (_Float16)a3[j];
            pb0[j] = (_Float16)b0[j]; pb0[j + 4] = (_Float16)b1[j];
            pb1[j] = (_Float16)b2[j]; pb1[j + 4] = (_Float16)b3[j];
        }
        *(h8*)&At[lrow * LDT + lhalf]     = pa0;
        *(h8*)&At[lrow * LDT + lhalf + 8] = pa1;
        *(h8*)&Bt[lrow * LDT + lhalf]     = pb0;
        *(h8*)&Bt[lrow * LDT + lhalf + 8] = pb1;
        __syncthreads();

        // ---- fragments per CDNA5 16-bit WMMA VGPR layouts ----
        Frag af[2], bf[4];
        #pragma unroll
        for (int i = 0; i < 2; ++i) {
            const int r  = wm + i * 16 + (lane & 15);
            const int ko = (lane >> 4) * 8;            // K 0-7 / 8-15 halves
            af[i].h[0] = *(const h8*)&At[r * LDT + ko];
            af[i].h[1] = *(const h8*)&At[r * LDT + ko + 16];
        }
        #pragma unroll
        for (int j = 0; j < 4; ++j) {
            const int c  = wn + j * 16 + (lane & 15);
            const int ko = (lane >> 4) * 8;            // K 0-7 / 8-15 halves
            bf[j].h[0] = *(const h8*)&Bt[c * LDT + ko];
            bf[j].h[1] = *(const h8*)&Bt[c * LDT + ko + 16];
        }
        #pragma unroll
        for (int i = 0; i < 2; ++i)
            #pragma unroll
            for (int j = 0; j < 4; ++j)
                acc[i][j] = __builtin_amdgcn_wmma_f32_16x16x32_f16(
                    false, af[i].v, false, bf[j].v, (short)0, acc[i][j], false, false);
    }

    // ---- epilogue: bias + sigmoid, guarded store ----
    #pragma unroll
    for (int i = 0; i < 2; ++i) {
        #pragma unroll
        for (int j = 0; j < 4; ++j) {
            const int col = bn + wn + j * 16 + (lane & 15);
            if (col < N_NODE) {
                const float bv   = bias[col];
                const int rbase  = bm + wm + i * 16 + (lane >> 4) * 8;
                #pragma unroll
                for (int r = 0; r < 8; ++r) {
                    const float x = acc[i][j][r] + bv;
                    ps[(size_t)(rbase + r) * N_PAD + col] = 1.0f / (1.0f + __expf(-x));
                }
            }
        }
    }
}

// ---------------------------------------------------------------------------
// Kernel 2: path probabilities pa[l] = prod over levels of p / (1-p), f16 out
// ---------------------------------------------------------------------------
__global__ __launch_bounds__(256)
void sdt_path_probs(const float* __restrict__ ps, _Float16* __restrict__ paH)
{
    __shared__ float prow[N_NODE + 1];
    const int row = blockIdx.x;
    const float* pr = ps + (size_t)row * N_PAD;
    for (int i = threadIdx.x; i < N_NODE; i += 256) prow[i] = pr[i];
    __syncthreads();

    for (int l = threadIdx.x; l < N_LEAF; l += 256) {
        float prob = 1.0f;
        #pragma unroll
        for (int t = 0; t < 10; ++t) {
            const int   k   = l >> (10 - t);                 // node within level t
            const float p   = prow[(1 << t) - 1 + k];
            const int   bit = (l >> (9 - t)) & 1;            // go right?
            prob *= bit ? p : (1.0f - p);
        }
        paH[(size_t)row * N_LEAF + l] = (_Float16)prob;
    }
}

// ---------------------------------------------------------------------------
// Kernel 3: dists_t[class][leaf] = softmax(leaf_logits[leaf])[class], f16,
// zero-filled for class in [1000,1024).
// ---------------------------------------------------------------------------
__global__ __launch_bounds__(256)
void sdt_softmax_t(const float* __restrict__ logits, _Float16* __restrict__ distsT)
{
    __shared__ float red[256];
    const int leaf = blockIdx.x;
    const float* lr = logits + (size_t)leaf * N_CLS;

    float m = -3.4e38f;
    for (int i = threadIdx.x; i < N_CLS; i += 256) m = fmaxf(m, lr[i]);
    red[threadIdx.x] = m;
    __syncthreads();
    for (int s = 128; s > 0; s >>= 1) {
        if (threadIdx.x < s) red[threadIdx.x] = fmaxf(red[threadIdx.x], red[threadIdx.x + s]);
        __syncthreads();
    }
    m = red[0];
    __syncthreads();

    float sum = 0.0f;
    for (int i = threadIdx.x; i < N_CLS; i += 256) sum += __expf(lr[i] - m);
    red[threadIdx.x] = sum;
    __syncthreads();
    for (int s = 128; s > 0; s >>= 1) {
        if (threadIdx.x < s) red[threadIdx.x] += red[threadIdx.x + s];
        __syncthreads();
    }
    const float inv = 1.0f / red[0];

    for (int n = threadIdx.x; n < N_PAD; n += 256) {
        const float v = (n < N_CLS) ? __expf(lr[n] - m) * inv : 0.0f;
        distsT[(size_t)n * N_LEAF + leaf] = (_Float16)v;
    }
}

// ---------------------------------------------------------------------------
// Kernel 4: out = pa @ dists   M=8192 N=1000(pad 1024) K=1024, f16 WMMA.
// Tiles staged via CDNA5 async global->LDS DMA (ASYNCcnt), LDS double-buffered:
// issue tile i+1, s_wait_asynccnt<=4 completes tile i (in-order), compute i.
// INST_OFFSET applies to both LDS and global address, so chunk 2 is offset:16.
// ---------------------------------------------------------------------------
__global__ __launch_bounds__(256)
void sdt_gemm2(const _Float16* __restrict__ paH, const _Float16* __restrict__ distsT,
               float* __restrict__ out)
{
    __shared__ _Float16 At[2][BM * LDT];
    __shared__ _Float16 Bt[2][BN * LDT];

    const int tid  = threadIdx.x;
    const int lane = tid & 31;
    const int wid  = tid >> 5;
    const int bm   = blockIdx.x * BM;
    const int bn   = blockIdx.y * BN;
    const int wm   = (wid & 3) * 32;
    const int wn   = (wid >> 2) * 64;

    const int lrow  = tid >> 1;        // tile row this thread stages
    const int lhalf = (tid & 1) * 16;  // K half it stages

    v8f acc[2][4] = {};

    const _Float16* aSrc = paH    + (size_t)(bm + lrow) * N_LEAF + lhalf;
    const _Float16* bSrc = distsT + (size_t)(bn + lrow) * N_LEAF + lhalf;

    // LDS byte addresses (low 32 bits of flat LDS pointer == LDS offset)
    const unsigned ldsA[2] = { (unsigned)(size_t)&At[0][lrow * LDT + lhalf],
                               (unsigned)(size_t)&At[1][lrow * LDT + lhalf] };
    const unsigned ldsB[2] = { (unsigned)(size_t)&Bt[0][lrow * LDT + lhalf],
                               (unsigned)(size_t)&Bt[1][lrow * LDT + lhalf] };

    auto issue_tile = [&](int buf, int k0) {
        const _Float16* ga = aSrc + k0;
        const _Float16* gb = bSrc + k0;
        asm volatile(
            "global_load_async_to_lds_b128 %0, %2, off\n\t"
            "global_load_async_to_lds_b128 %0, %2, off offset:16\n\t"
            "global_load_async_to_lds_b128 %1, %3, off\n\t"
            "global_load_async_to_lds_b128 %1, %3, off offset:16\n\t"
            :: "v"(ldsA[buf]), "v"(ldsB[buf]), "v"(ga), "v"(gb)
            : "memory");
    };

    constexpr int NK = N_LEAF / BK;    // 32 k-steps
    issue_tile(0, 0);                  // prologue

    for (int i = 0; i < NK; ++i) {
        const int buf = i & 1;
        if (i + 1 < NK) {
            issue_tile(buf ^ 1, (i + 1) * BK);
            asm volatile("s_wait_asynccnt 4" ::: "memory");  // tile i landed
        } else {
            asm volatile("s_wait_asynccnt 0" ::: "memory");
        }
        __syncthreads();               // tile i visible to all waves

        const _Float16* Ab = At[buf];
        const _Float16* Bb = Bt[buf];
        Frag af[2], bf[4];
        #pragma unroll
        for (int ii = 0; ii < 2; ++ii) {
            const int r  = wm + ii * 16 + (lane & 15);
            const int ko = (lane >> 4) * 8;
            af[ii].h[0] = *(const h8*)&Ab[r * LDT + ko];
            af[ii].h[1] = *(const h8*)&Ab[r * LDT + ko + 16];
        }
        #pragma unroll
        for (int j = 0; j < 4; ++j) {
            const int c  = wn + j * 16 + (lane & 15);
            const int ko = (lane >> 4) * 8;
            bf[j].h[0] = *(const h8*)&Bb[c * LDT + ko];
            bf[j].h[1] = *(const h8*)&Bb[c * LDT + ko + 16];
        }
        #pragma unroll
        for (int ii = 0; ii < 2; ++ii)
            #pragma unroll
            for (int j = 0; j < 4; ++j)
                acc[ii][j] = __builtin_amdgcn_wmma_f32_16x16x32_f16(
                    false, af[ii].v, false, bf[j].v, (short)0, acc[ii][j], false, false);

        __syncthreads();               // done reading buf before it is re-filled
    }

    #pragma unroll
    for (int i = 0; i < 2; ++i) {
        #pragma unroll
        for (int j = 0; j < 4; ++j) {
            const int col = bn + wn + j * 16 + (lane & 15);
            if (col < N_CLS) {
                const int rbase = bm + wm + i * 16 + (lane >> 4) * 8;
                #pragma unroll
                for (int r = 0; r < 8; ++r)
                    out[(size_t)(rbase + r) * N_CLS + col] = acc[i][j][r];
            }
        }
    }
}

// ---------------------------------------------------------------------------
extern "C" void kernel_launch(void* const* d_in, const int* in_sizes, int n_in,
                              void* d_out, int out_size, void* d_ws, size_t ws_size,
                              hipStream_t stream)
{
    const float* xs     = (const float*)d_in[0];   // 8192 x 3072
    const float* W      = (const float*)d_in[1];   // 1023 x 3072
    const float* bias   = (const float*)d_in[2];   // 1023
    const float* logits = (const float*)d_in[3];   // 1024 x 1000
    float*       out    = (float*)d_out;           // 8192 x 1000

    char* ws = (char*)d_ws;
    const size_t PS_BYTES = (size_t)BATCH_N * N_PAD * sizeof(float);     // 32 MiB
    const size_t PA_BYTES = (size_t)BATCH_N * N_LEAF * sizeof(_Float16); // 16 MiB
    float*    ps     = (float*)ws;
    _Float16* paH    = (_Float16*)(ws + PS_BYTES);
    _Float16* distsT = (_Float16*)(ws + PS_BYTES + PA_BYTES);            // 2 MiB

    (void)in_sizes; (void)n_in; (void)out_size; (void)ws_size;

    // independent of gemm1; runs first on the stream
    sdt_softmax_t<<<N_LEAF, 256, 0, stream>>>(logits, distsT);

    sdt_gemm1_sigmoid<<<dim3(BATCH_N / BM, N_PAD / BN), 256, 0, stream>>>(xs, W, bias, ps);

    sdt_path_probs<<<BATCH_N, 256, 0, stream>>>(ps, paH);

    sdt_gemm2<<<dim3(BATCH_N / BM, N_PAD / BN), 256, 0, stream>>>(paH, distsT, out);
}
